// ACLSTM_15796889715035
// MI455X (gfx1250) — matrix operand order repaired and checked by reading, not compiled, not run on hardware
//
#include <hip/hip_runtime.h>
#include <hip/hip_bf16.h>

typedef __bf16 bf16_t;
typedef __attribute__((ext_vector_type(16))) __bf16 v16bf;
typedef __attribute__((ext_vector_type(8)))  __bf16 v8bf;
typedef __attribute__((ext_vector_type(8)))  float  v8f;

#define N_SEQ   256
#define T_STEPS 512
#define IN_F    128
#define H_DIM   256
#define A_DIM   32
#define G4      1024   // 4*H

// workspace layout (bytes)
#define WS_WIH   0                          // 1024*128 bf16 = 256 KB
#define WS_WHH   (WS_WIH + G4*IN_F*2)       // 1024*256 bf16 = 512 KB
#define WS_WHEAD (WS_WHH + G4*H_DIM*2)      // 48*256  bf16 =  24 KB
#define WS_BIAS  (WS_WHEAD + 48*H_DIM*2)    // 1024 f32
#define WS_BHEAD (WS_BIAS + G4*4)           // 48 f32

// output layout (f32 elements): logits | value | h | c
#define OUT_LOGITS 0
#define OUT_VALUE  (N_SEQ*T_STEPS*A_DIM)
#define OUT_H      (OUT_VALUE + N_SEQ*T_STEPS)
#define OUT_C      (OUT_H + N_SEQ*H_DIM)

// ---- WMMA fragment helpers (contiguous-K order k = 16*(lane/16) + s, 16-bit, wave32) ----
// A (16x32, MxK): lane holds row m=lane&15; 16 consecutive K values [k0..k0+15],
// k0 = ktile*32 + (lane&16 ? 16 : 0).  Two ds_load_b128 per fragment.
__device__ __forceinline__ v16bf load_a_frag(const bf16_t* base, int stride,
                                             int ktile, int lane) {
  const int m  = lane & 15;
  const int k0 = ktile * 32 + ((lane & 16) ? 16 : 0);
  const bf16_t* p = base + m * stride + k0;
  v8bf lo = *(const v8bf*)(p);
  v8bf hi = *(const v8bf*)(p + 8);
  v16bf a;
#pragma unroll
  for (int i = 0; i < 8; ++i) { a[i] = lo[i]; a[i + 8] = hi[i]; }
  return a;
}

// B (32x16, KxN): lane holds column n=lane&15; 16 consecutive K values starting
// at ktile*32 + (lane&16 ? 16 : 0). With gates = a @ W^T, B[k][n] = W[n][k]:
// 32 contiguous bytes from row n of the row-major bf16 weight matrix.
// Address = uniform base (SADDR) + loop-invariant lane offset (voffset) +
// compile-time k-tile displacement (IOFFSET).
__device__ __forceinline__ v16bf load_b_at(const bf16_t* base_plus_vofs, int ktile) {
  return *(const v16bf*)(base_plus_vofs + ktile * 32);
}

__device__ __forceinline__ float sigmoidf_(float x) {
  return 1.0f / (1.0f + __expf(-x));
}

#define WMMA_BF16(A, B, C) \
  __builtin_amdgcn_wmma_f32_16x16x32_bf16(false, (A), false, (B), (short)0, (C), false, false)

// ---- prep: f32 weights -> bf16 workspace, fold biases, pack head matrix ----
__global__ void aclstm_prep(const float* __restrict__ W_ih, const float* __restrict__ W_hh,
                            const float* __restrict__ b_ih, const float* __restrict__ b_hh,
                            const float* __restrict__ W_logit, const float* __restrict__ b_logit,
                            const float* __restrict__ W_value, const float* __restrict__ b_value,
                            bf16_t* __restrict__ wih, bf16_t* __restrict__ whh,
                            bf16_t* __restrict__ whead, float* __restrict__ bias,
                            float* __restrict__ bhead) {
  const int total = G4 * H_DIM;  // largest array: W_hh (262144)
  for (int i = blockIdx.x * blockDim.x + threadIdx.x; i < total;
       i += gridDim.x * blockDim.x) {
    whh[i] = (bf16_t)W_hh[i];
    if (i < G4 * IN_F) wih[i] = (bf16_t)W_ih[i];
    if (i < 48 * H_DIM) {
      int n = i >> 8, k = i & 255;
      float v = (n < A_DIM) ? W_logit[n * H_DIM + k]
                            : ((n == A_DIM) ? W_value[k] : 0.0f);
      whead[i] = (bf16_t)v;
    }
    if (i < G4) bias[i] = b_ih[i] + b_hh[i];
    if (i < 48) bhead[i] = (i < A_DIM) ? b_logit[i]
                                       : ((i == A_DIM) ? b_value[0] : 0.0f);
  }
}

// ---- persistent fused LSTM: 16 blocks x 16 sequences, 512 steps in-kernel ----
__global__ __launch_bounds__(512)
void aclstm_recur(const float* __restrict__ features,
                  const float* __restrict__ starts,
                  const float* __restrict__ h0,
                  const float* __restrict__ c0,
                  const bf16_t* __restrict__ wih,
                  const bf16_t* __restrict__ whh,
                  const bf16_t* __restrict__ whead,
                  const float* __restrict__ bias,
                  const float* __restrict__ bhead,
                  float* __restrict__ out) {
  __shared__ bf16_t xb[16][IN_F];      //  4 KB  current x tile (bf16)
  __shared__ bf16_t hb[2][16][H_DIM];  // 16 KB  h (bf16, double-buffered A operand)
  __shared__ float  hf[16][H_DIM];     // 16 KB  h (f32, exact final output)
  __shared__ float  cf[16][H_DIM];     // 16 KB  c (f32, recurrence precision)

  const int tid  = threadIdx.x;
  const int lane = tid & 31;
  const int wave = tid >> 5;               // 16 waves
  const int seq0 = blockIdx.x * 16;

  // init h, c from h0/c0 (h_0 lives in hb[0])
  {
    int idx = tid * 8;                     // 512*8 = 4096 = 16*256
    int m = idx >> 8, k = idx & 255;
#pragma unroll
    for (int i = 0; i < 8; ++i) {
      float hv = h0[(size_t)(seq0 + m) * H_DIM + k + i];
      float cv = c0[(size_t)(seq0 + m) * H_DIM + k + i];
      hf[m][k + i] = hv; hb[0][m][k + i] = (bf16_t)hv; cf[m][k + i] = cv;
    }
  }

  // ---- per-lane loop-invariant constants ----
  const int   jloc = lane & 15;
  const int   j    = wave * 16 + jloc;     // hidden column owned by this lane
  const float b_i  = bias[j];
  const float b_f  = bias[256 + j];
  const float b_g  = bias[512 + j];
  const float b_o  = bias[768 + j];
  const int   roff = (lane & 16) ? 8 : 0;  // C-fragment row offset
  const float bh   = (wave < 3) ? bhead[wave * 16 + jloc] : 0.0f;
  const int   kb_b = (lane & 16) ? 16 : 0; // B-fragment K sub-offset

  // per-lane element offsets into the weight matrices (loop-invariant, VGPR)
  const int vo_ih_i = ((wave +  0) * 16 + jloc) * IN_F  + kb_b;
  const int vo_ih_f = ((wave + 16) * 16 + jloc) * IN_F  + kb_b;
  const int vo_ih_g = ((wave + 32) * 16 + jloc) * IN_F  + kb_b;
  const int vo_ih_o = ((wave + 48) * 16 + jloc) * IN_F  + kb_b;
  const int vo_hh_i = ((wave +  0) * 16 + jloc) * H_DIM + kb_b;
  const int vo_hh_f = ((wave + 16) * 16 + jloc) * H_DIM + kb_b;
  const int vo_hh_g = ((wave + 32) * 16 + jloc) * H_DIM + kb_b;
  const int vo_hh_o = ((wave + 48) * 16 + jloc) * H_DIM + kb_b;
  const int vo_hd   = (wave * 16 + jloc) * H_DIM + kb_b;

  // staging offsets (per thread, loop-invariant)
  const int sm_x = (tid * 4) >> 7,  sk_x = (tid * 4) & 127;   // x stage
  const int fvo  = sm_x * T_STEPS * IN_F + sk_x;
  const int sm_r = tid >> 5,        sk_r = (tid & 31) * 8;    // reset pass
  const int svo  = sm_r * T_STEPS;
  const int vbL  = roff * T_STEPS * A_DIM + wave * 16 + jloc; // logits store
  const int vbV  = roff * T_STEPS;                            // value store

  // uniform stepping base pointers (SGPR; one scalar add per step each)
  const float* fbase = features + (size_t)seq0 * T_STEPS * IN_F;
  const float* sbase = starts   + (size_t)seq0 * T_STEPS;
  float*       lbase = out + OUT_LOGITS + (size_t)seq0 * T_STEPS * A_DIM;
  float*       vbase = out + OUT_VALUE  + (size_t)seq0 * T_STEPS;

  __syncthreads();

  for (int t = 0; t < T_STEPS; ++t) {
    // Opaque scalar offset: keeps the per-step B loads inside the loop
    // (streamed from the 192MB L2) without breaking global addrspace
    // inference -> global_load_b128 saddr+voffset, no LICM spill.
    int opq = 0;
    asm volatile("" : "+s"(opq));
    const bf16_t* pih_i = wih + opq + vo_ih_i;
    const bf16_t* pih_f = wih + opq + vo_ih_f;
    const bf16_t* pih_g = wih + opq + vo_ih_g;
    const bf16_t* pih_o = wih + opq + vo_ih_o;
    const bf16_t* phh_i = whh + opq + vo_hh_i;
    const bf16_t* phh_f = whh + opq + vo_hh_f;
    const bf16_t* phh_g = whh + opq + vo_hh_g;
    const bf16_t* phh_o = whh + opq + vo_hh_o;
    const bf16_t* phd   = whead + opq + vo_hd;

    const int p = t & 1;                   // matmul reads hb[p], writes hb[p^1]

    // 1) stage x_t tile into LDS (f32 -> bf16)
    {
      const float4 xv = *(const float4*)(fbase + fvo);
      xb[sm_x][sk_x]     = (bf16_t)xv.x;
      xb[sm_x][sk_x + 1] = (bf16_t)xv.y;
      xb[sm_x][sk_x + 2] = (bf16_t)xv.z;
      xb[sm_x][sk_x + 3] = (bf16_t)xv.w;
    }
    // 2) episode reset: h,c *= (1 - start)  (applies to hb[p]/hf/cf)
    {
      float st = sbase[svo];
      if (st != 0.0f) {
#pragma unroll
        for (int i = 0; i < 8; ++i) {
          hf[sm_r][sk_r + i] = 0.0f;
          cf[sm_r][sk_r + i] = 0.0f;
          hb[p][sm_r][sk_r + i] = (bf16_t)0.0f;
        }
      }
    }
    __syncthreads();   // B_a

    // 3) gates = [x | h] @ [W_ih | W_hh]^T, software-pipelined B fetch.
    //    kt 0..3: x (K=128, wih); kt 4..11: h (K=256, whh).
    v8f acc_i = {}, acc_f = {}, acc_g = {}, acc_o = {};
    {
      auto loadB4 = [&](int kt, v16bf& bi, v16bf& bff, v16bf& bg, v16bf& bo) {
        if (kt < 4) {
          bi  = load_b_at(pih_i, kt);
          bff = load_b_at(pih_f, kt);
          bg  = load_b_at(pih_g, kt);
          bo  = load_b_at(pih_o, kt);
        } else {
          bi  = load_b_at(phh_i, kt - 4);
          bff = load_b_at(phh_f, kt - 4);
          bg  = load_b_at(phh_g, kt - 4);
          bo  = load_b_at(phh_o, kt - 4);
        }
      };
      v16bf bi, bff, bg, bo;
      loadB4(0, bi, bff, bg, bo);
#pragma unroll
      for (int kt = 0; kt < 12; ++kt) {
        v16bf a = (kt < 4) ? load_a_frag(&xb[0][0], IN_F, kt, lane)
                           : load_a_frag(&hb[p][0][0], H_DIM, kt - 4, lane);
        v16bf ni, nf, ng, no;
        if (kt < 11) loadB4(kt + 1, ni, nf, ng, no);   // prefetch next group
        acc_i = WMMA_BF16(a, bi,  acc_i);
        acc_f = WMMA_BF16(a, bff, acc_f);
        acc_g = WMMA_BF16(a, bg,  acc_g);
        acc_o = WMMA_BF16(a, bo,  acc_o);
        bi = ni; bff = nf; bg = ng; bo = no;
      }
    }

    // 4) LSTM pointwise update in registers; write h into the OTHER hb buffer
#pragma unroll
    for (int r = 0; r < 8; ++r) {
      int m = r + roff;
      float iv = sigmoidf_(acc_i[r] + b_i);
      float fv = sigmoidf_(acc_f[r] + b_f);
      float gv = tanhf(acc_g[r] + b_g);
      float ov = sigmoidf_(acc_o[r] + b_o);
      float cn = fv * cf[m][j] + iv * gv;
      float hn = ov * tanhf(cn);
      cf[m][j] = cn;
      hf[m][j] = hn;
      hb[p ^ 1][m][j] = (bf16_t)hn;
    }
    __syncthreads();   // B_b: fresh h visible for heads / next-step matmul

    // 5) heads: logits (tiles 0,1) + value (tile 2, col 32) from h_t = hb[p^1]
    if (wave < 3) {                        // wave-uniform branch: EXEC stays full
      v8f acc = {};
      v16bf b = load_b_at(phd, 0);
#pragma unroll
      for (int kt = 0; kt < 8; ++kt) {
        v16bf a = load_a_frag(&hb[p ^ 1][0][0], H_DIM, kt, lane);
        v16bf nb;
        if (kt < 7) nb = load_b_at(phd, kt + 1);
        acc = WMMA_BF16(a, b, acc);
        b = nb;
      }
#pragma unroll
      for (int r = 0; r < 8; ++r) {
        float v = acc[r] + bh;
        if (wave < 2)          lbase[vbL + r * T_STEPS * A_DIM] = v;  // r-disp -> IOFFSET
        else if (jloc == 0)    vbase[vbV + r * T_STEPS] = v;
      }
    }
    __syncthreads();   // B_c: heads done with hb[p^1] before next-step reset/stage

    // advance uniform stepping bases (scalar adds)
    fbase += IN_F;
    sbase += 1;
    lbase += A_DIM;
    vbase += 1;
  }

  // final h, c outputs (f32)
  {
    int idx = tid * 8;
    int m = idx >> 8, k = idx & 255;
#pragma unroll
    for (int i = 0; i < 8; ++i) {
      out[OUT_H + (size_t)(seq0 + m) * H_DIM + k + i] = hf[m][k + i];
      out[OUT_C + (size_t)(seq0 + m) * H_DIM + k + i] = cf[m][k + i];
    }
  }
}

extern "C" void kernel_launch(void* const* d_in, const int* in_sizes, int n_in,
                              void* d_out, int out_size, void* d_ws, size_t ws_size,
                              hipStream_t stream) {
  (void)in_sizes; (void)n_in; (void)out_size; (void)ws_size;
  const float* features = (const float*)d_in[0];
  const float* starts   = (const float*)d_in[1];
  const float* h0       = (const float*)d_in[2];
  const float* c0       = (const float*)d_in[3];
  const float* W_ih     = (const float*)d_in[4];
  const float* W_hh     = (const float*)d_in[5];
  const float* b_ih     = (const float*)d_in[6];
  const float* b_hh     = (const float*)d_in[7];
  const float* W_logit  = (const float*)d_in[8];
  const float* b_logit  = (const float*)d_in[9];
  const float* W_value  = (const float*)d_in[10];
  const float* b_value  = (const float*)d_in[11];

  char* ws = (char*)d_ws;
  bf16_t* wih   = (bf16_t*)(ws + WS_WIH);
  bf16_t* whh   = (bf16_t*)(ws + WS_WHH);
  bf16_t* whead = (bf16_t*)(ws + WS_WHEAD);
  float*  bias  = (float*)(ws + WS_BIAS);
  float*  bhead = (float*)(ws + WS_BHEAD);

  aclstm_prep<<<256, 256, 0, stream>>>(W_ih, W_hh, b_ih, b_hh,
                                       W_logit, b_logit, W_value, b_value,
                                       wih, whh, whead, bias, bhead);
  aclstm_recur<<<N_SEQ / 16, 512, 0, stream>>>(features, starts, h0, c0,
                                               wih, whh, whead, bias, bhead,
                                               (float*)d_out);
}
